// TransformerEncoder_47399259079089
// MI455X (gfx1250) — matrix operand order repaired and checked by reading, not compiled, not run on hardware
//
#include <hip/hip_runtime.h>
#include <hip/hip_bf16.h>
#include <stdint.h>

// Transformer encoder (GPT2-base shape) for gfx1250 via bf16 WMMA,
// async global->LDS double-buffered GEMM tiles.
#define LNUM   12
#define DMODEL 768
#define NHEAD  12
#define DHEAD  64
#define DFFN   3072
#define BATCHN 8
#define SEQ    512
#define NTOK   (BATCHN * SEQ)   // 4096
#define EPSLN  1e-5f

typedef __attribute__((ext_vector_type(16))) __bf16 v16bf;
typedef __attribute__((ext_vector_type(8)))  float  v8f;
typedef int gv4i __attribute__((vector_size(16)));   // matches builtin's int4

__device__ __forceinline__ unsigned short f2bf(float f) {
  unsigned u = __float_as_uint(f);
  u += 0x7FFFu + ((u >> 16) & 1u);          // round-to-nearest-even
  return (unsigned short)(u >> 16);
}
__device__ __forceinline__ float bf2f(unsigned short s) {
  return __uint_as_float(((unsigned)s) << 16);
}

// 16-byte global -> LDS copy. Prefer CDNA5 async-to-LDS (ASYNCcnt-tracked,
// no VGPR round trip); fall back to a sync copy if the builtin is absent.
__device__ __forceinline__ void cp16_g2l(const unsigned short* g, unsigned short* l) {
#if defined(__gfx1250__) && __has_builtin(__builtin_amdgcn_global_load_async_to_lds_b128)
  gv4i* gp = reinterpret_cast<gv4i*>(const_cast<unsigned short*>(g));
  gv4i* lp = reinterpret_cast<gv4i*>(l);
  __builtin_amdgcn_global_load_async_to_lds_b128(
      (__attribute__((address_space(1))) gv4i*)gp,
      (__attribute__((address_space(3))) gv4i*)lp, 0, 0);
#else
  *(uint4*)l = *(const uint4*)g;
#endif
}
__device__ __forceinline__ void wait_async_all() {
#if defined(__gfx1250__) && __has_builtin(__builtin_amdgcn_s_wait_asynccnt)
  __builtin_amdgcn_s_wait_asynccnt(0);
#elif defined(__gfx1250__)
  asm volatile("s_wait_asynccnt 0" ::: "memory");
#endif
}

// ---------------------------------------------------------------- converters
__global__ void cvt_f32_bf16(const float* __restrict__ src,
                             unsigned short* __restrict__ dst, int n) {
  for (int i = blockIdx.x * blockDim.x + threadIdx.x; i < n;
       i += gridDim.x * blockDim.x)
    dst[i] = f2bf(src[i]);
}

// ---------------------------------------------------------------- GEMM (NT)
// C[M,N] = epilogue( A[M,K] @ B[N,K]^T * scale + bias[n] )
// A,B bf16 row-major (K contiguous). 256 threads = 8 waves; block tile
// 128x64, each wave owns a 16x64 strip (4 f32 accumulators), BK=64,
// double-buffered LDS tiles filled with async copies.
#define BM  128
#define BN  64
#define BK  64
#define LDT 72   // padded LDS row stride (bf16 elems), 144B = 16B aligned

#define OUT_BF16 0
#define OUT_F32  1
#define OUT_QK   2   // m=(b,s), n=(h,dh) -> [b,h,s,dh] bf16
#define OUT_VT   3   // m=(b,s), n=(h,dh) -> [b,h,dh,s] bf16
#define OUT_CTX  4   // z=(b,h), m=q, n=dh -> [b,q,h*DH+dh] bf16

__global__ __launch_bounds__(256, 2)
void gemm_nt_bf16(const unsigned short* __restrict__ A, long long aBatch, int lda,
                  const unsigned short* __restrict__ B, long long bBatch, int ldb,
                  const float* __restrict__ bias, float scale, int relu,
                  int K, int mode, void* __restrict__ Cv, long long cBatch, int ldc)
{
  __shared__ __align__(16) unsigned short lA[2][BM * LDT];
  __shared__ __align__(16) unsigned short lB[2][BN * LDT];

  const int tid  = threadIdx.x;
  const int wid  = tid >> 5;
  const int lane = tid & 31;
  const int bm   = blockIdx.y * BM;
  const int bn   = blockIdx.x * BN;

  // per-thread staging geometry: thread t owns row (t>>3)+32*i, 16B chunk t&7
  const int srow = tid >> 3;            // 0..31
  const int skp  = (tid & 7) * 8;       // bf16 element offset 0..56
  const unsigned short* gA = A + (long long)blockIdx.z * aBatch +
                             (size_t)(bm + srow) * lda + skp;
  const unsigned short* gB = B + (long long)blockIdx.z * bBatch +
                             (size_t)(bn + srow) * ldb + skp;
  const size_t sa32 = (size_t)32 * lda;
  const size_t sb32 = (size_t)32 * ldb;
  const int lofsA = srow * LDT + skp;
  const int lofsB = srow * LDT + skp;

  v8f acc[4];
  #pragma unroll
  for (int i = 0; i < 4; ++i) acc[i] = (v8f){0.f,0.f,0.f,0.f,0.f,0.f,0.f,0.f};

  // prologue: stage tile 0 into buffer 0
  #pragma unroll
  for (int i = 0; i < 4; ++i) cp16_g2l(gA + i * sa32, &lA[0][lofsA + i * 32 * LDT]);
  #pragma unroll
  for (int i = 0; i < 2; ++i) cp16_g2l(gB + i * sb32, &lB[0][lofsB + i * 32 * LDT]);
  wait_async_all();
  __syncthreads();

  const int nk = K / BK;
  const int arow = wid * 16 + (lane & 15);
  const int ka0  = (lane < 16) ? 0 : 8;    // A frag: K {0..7,16..23} | {8..15,24..31}
  const int kb0  = (lane < 16) ? 0 : 8;    // B frag: K {0..7,16..23} | {8..15,24..31}

  for (int t = 0; t < nk; ++t) {
    const int cur = t & 1;
    if (t + 1 < nk) {          // async-stage next tile into the other buffer
      const int k1 = (t + 1) * BK;
      #pragma unroll
      for (int i = 0; i < 4; ++i)
        cp16_g2l(gA + k1 + i * sa32, &lA[cur ^ 1][lofsA + i * 32 * LDT]);
      #pragma unroll
      for (int i = 0; i < 2; ++i)
        cp16_g2l(gB + k1 + i * sb32, &lB[cur ^ 1][lofsB + i * 32 * LDT]);
    }

    // hoist ALL fragments, then a dense burst of 8 WMMAs (single dscnt wait)
    union Frag { uint4 u[2]; v16bf v; };
    Frag fa[2], fb[2][4];
    const unsigned short* La = lA[cur];
    const unsigned short* Lb = lB[cur];
    #pragma unroll
    for (int kk = 0; kk < 2; ++kk) {
      fa[kk].u[0] = *(const uint4*)(La + arow * LDT + kk * 32 + ka0);
      fa[kk].u[1] = *(const uint4*)(La + arow * LDT + kk * 32 + ka0 + 16);
      #pragma unroll
      for (int nt = 0; nt < 4; ++nt) {
        const int brow = nt * 16 + (lane & 15);
        fb[kk][nt].u[0] = *(const uint4*)(Lb + brow * LDT + kk * 32 + kb0);
        fb[kk][nt].u[1] = *(const uint4*)(Lb + brow * LDT + kk * 32 + kb0 + 16);
      }
    }
    #pragma unroll
    for (int kk = 0; kk < 2; ++kk)
      #pragma unroll
      for (int nt = 0; nt < 4; ++nt)
        acc[nt] = __builtin_amdgcn_wmma_f32_16x16x32_bf16(
            false, fa[kk].v, false, fb[kk][nt].v, (short)0, acc[nt], false, false);

    wait_async_all();      // our next-tile async copies have landed
    __syncthreads();       // everyone's have; safe to flip buffers
  }

  // epilogue: VGPR r holds (M=r | r+8, N=lane&15) per ISA C/D layout
  const int mbase = bm + wid * 16 + ((lane >> 4) << 3);
  const int nlane = lane & 15;
  #pragma unroll
  for (int nt = 0; nt < 4; ++nt) {
    #pragma unroll
    for (int r = 0; r < 8; ++r) {
      const int m = mbase + r;
      const int n = bn + nt * 16 + nlane;
      float v = acc[nt][r] * scale;
      if (bias) v += bias[n];
      if (relu) v = fmaxf(v, 0.0f);
      if (mode == OUT_BF16) {
        ((unsigned short*)Cv)[(size_t)blockIdx.z * cBatch + (size_t)m * ldc + n] = f2bf(v);
      } else if (mode == OUT_F32) {
        ((float*)Cv)[(size_t)blockIdx.z * cBatch + (size_t)m * ldc + n] = v;
      } else if (mode == OUT_QK) {
        const int b = m / SEQ, s = m % SEQ, hh = n / DHEAD, dh = n % DHEAD;
        ((unsigned short*)Cv)[(((size_t)(b * NHEAD + hh) * SEQ + s) * DHEAD) + dh] = f2bf(v);
      } else if (mode == OUT_VT) {
        const int b = m / SEQ, s = m % SEQ, hh = n / DHEAD, dh = n % DHEAD;
        ((unsigned short*)Cv)[(((size_t)(b * NHEAD + hh) * DHEAD + dh) * SEQ) + s] = f2bf(v);
      } else { // OUT_CTX
        const int bh = blockIdx.z, b = bh / NHEAD, hh = bh % NHEAD;
        ((unsigned short*)Cv)[((size_t)b * SEQ + m) * DMODEL + hh * DHEAD + n] = f2bf(v);
      }
    }
  }
}

// ---------------------------------------------------------------- softmax
// one wave32 per row of 512 scores; scale applied in GEMM epilogue.
__global__ __launch_bounds__(128)
void softmax_rows(unsigned short* __restrict__ sc, const float* __restrict__ mask)
{
  const int lane = threadIdx.x & 31;
  const int wid  = threadIdx.x >> 5;
  const int row  = blockIdx.x * 4 + wid;   // [0, 96*512)
  const int bh   = row >> 9;
  const int b    = bh / NHEAD;
  unsigned short* p = sc + (size_t)row * SEQ + lane * 16;
  const float* mk   = mask + (size_t)b * SEQ + lane * 16;

  union { uint4 u[2]; unsigned short s[16]; } d;
  d.u[0] = *(const uint4*)(p);
  d.u[1] = *(const uint4*)(p + 8);
  float x[16];
  float mx = -3.0e38f;
  #pragma unroll
  for (int j = 0; j < 16; ++j) { x[j] = bf2f(d.s[j]) + mk[j] * -1e9f; mx = fmaxf(mx, x[j]); }
  #pragma unroll
  for (int o = 16; o; o >>= 1) mx = fmaxf(mx, __shfl_xor(mx, o, 32));
  float sum = 0.f;
  #pragma unroll
  for (int j = 0; j < 16; ++j) { x[j] = __expf(x[j] - mx); sum += x[j]; }
  #pragma unroll
  for (int o = 16; o; o >>= 1) sum += __shfl_xor(sum, o, 32);
  const float inv = 1.0f / sum;
  #pragma unroll
  for (int j = 0; j < 16; ++j) d.s[j] = f2bf(x[j] * inv);
  *(uint4*)(p)     = d.u[0];
  *(uint4*)(p + 8) = d.u[1];
}

// ---------------------------------------------------------------- LN fused
// y = g*(r+d - mean)/sqrt(var+eps) + b ; also bf16 copy for next GEMM input.
__global__ __launch_bounds__(128)
void add_layernorm(const float* __restrict__ resid, const float* __restrict__ delta,
                   const float* __restrict__ g, const float* __restrict__ bb,
                   float* __restrict__ y, unsigned short* __restrict__ ybf)
{
  const int lane = threadIdx.x & 31;
  const int wid  = threadIdx.x >> 5;
  const int row  = blockIdx.x * 4 + wid;
  const size_t base = (size_t)row * DMODEL;
  float v[24];
  float s = 0.f;
  #pragma unroll
  for (int j = 0; j < 24; ++j) {
    const int c = j * 32 + lane;
    v[j] = resid[base + c] + delta[base + c];
    s += v[j];
  }
  #pragma unroll
  for (int o = 16; o; o >>= 1) s += __shfl_xor(s, o, 32);
  const float mean = s * (1.0f / DMODEL);
  float var = 0.f;
  #pragma unroll
  for (int j = 0; j < 24; ++j) { const float d = v[j] - mean; var += d * d; }
  #pragma unroll
  for (int o = 16; o; o >>= 1) var += __shfl_xor(var, o, 32);
  const float rstd = rsqrtf(var * (1.0f / DMODEL) + EPSLN);
  #pragma unroll
  for (int j = 0; j < 24; ++j) {
    const int c = j * 32 + lane;
    const float o2 = g[c] * (v[j] - mean) * rstd + bb[c];
    y[base + c]   = o2;
    ybf[base + c] = f2bf(o2);
  }
}

// ---------------------------------------------------------------- launcher
extern "C" void kernel_launch(void* const* d_in, const int* in_sizes, int n_in,
                              void* d_out, int out_size, void* d_ws, size_t ws_size,
                              hipStream_t stream)
{
  (void)in_sizes; (void)n_in; (void)out_size; (void)ws_size;
  const float* x    = (const float*)d_in[0];
  const float* mask = (const float*)d_in[1];
  const float* Wq = (const float*)d_in[2];   const float* bq  = (const float*)d_in[3];
  const float* Wk = (const float*)d_in[4];   const float* bk  = (const float*)d_in[5];
  const float* Wv = (const float*)d_in[6];   const float* bv  = (const float*)d_in[7];
  const float* Wo = (const float*)d_in[8];   const float* bo  = (const float*)d_in[9];
  const float* g1 = (const float*)d_in[10];  const float* b1  = (const float*)d_in[11];
  const float* W1 = (const float*)d_in[12];  const float* bf1 = (const float*)d_in[13];
  const float* W2 = (const float*)d_in[14];  const float* bf2 = (const float*)d_in[15];
  const float* g2 = (const float*)d_in[16];  const float* b2  = (const float*)d_in[17];

  char* ws = (char*)d_ws;
  size_t off = 0;
  auto alloc = [&](size_t bytes) -> char* {
    char* p = ws + off;
    off = (off + bytes + 255) & ~(size_t)255;
    return p;
  };

  unsigned short* wqb = (unsigned short*)alloc((size_t)DMODEL * DMODEL * 2);
  unsigned short* wkb = (unsigned short*)alloc((size_t)DMODEL * DMODEL * 2);
  unsigned short* wvb = (unsigned short*)alloc((size_t)DMODEL * DMODEL * 2);
  unsigned short* wob = (unsigned short*)alloc((size_t)DMODEL * DMODEL * 2);
  unsigned short* w1b = (unsigned short*)alloc((size_t)DFFN * DMODEL * 2);
  unsigned short* w2b = (unsigned short*)alloc((size_t)DFFN * DMODEL * 2);
  unsigned short* hbf = (unsigned short*)alloc((size_t)NTOK * DMODEL * 2);
  float*          h32 = (float*)alloc((size_t)NTOK * DMODEL * 4);
  unsigned short* qb  = (unsigned short*)alloc((size_t)NTOK * DMODEL * 2);
  unsigned short* kb2 = (unsigned short*)alloc((size_t)NTOK * DMODEL * 2);
  unsigned short* vtb = (unsigned short*)alloc((size_t)NTOK * DMODEL * 2);
  unsigned short* sc  = (unsigned short*)alloc((size_t)BATCHN * NHEAD * SEQ * SEQ * 2);
  unsigned short* ctxb= (unsigned short*)alloc((size_t)NTOK * DMODEL * 2);
  float*          t32 = (float*)alloc((size_t)NTOK * DMODEL * 4);
  float*          a32 = (float*)alloc((size_t)NTOK * DMODEL * 4);
  unsigned short* abf = (unsigned short*)alloc((size_t)NTOK * DMODEL * 2);
  unsigned short* f1b = (unsigned short*)alloc((size_t)NTOK * DFFN * 2);

  // h = x  (fp32 master + bf16 GEMM feed)
  (void)hipMemcpyAsync(h32, x, (size_t)NTOK * DMODEL * 4, hipMemcpyDeviceToDevice, stream);
  cvt_f32_bf16<<<4096, 256, 0, stream>>>(x, hbf, NTOK * DMODEL);

  const dim3 TB(256);
  const dim3 gProj(DMODEL / BN, NTOK / BM, 1);          // 12 x 32
  const dim3 gFfn1(DFFN / BN,  NTOK / BM, 1);           // 48 x 32
  const dim3 gScr(SEQ / BN, SEQ / BM, BATCHN * NHEAD);  // 8 x 4 x 96
  const dim3 gCtx(1, SEQ / BM, BATCHN * NHEAD);         // 1 x 4 x 96

  for (int l = 0; l < LNUM; ++l) {
    const size_t wD = (size_t)l * DMODEL * DMODEL;
    const size_t wF = (size_t)l * DFFN * DMODEL;
    cvt_f32_bf16<<<2304, 256, 0, stream>>>(Wq + wD, wqb, DMODEL * DMODEL);
    cvt_f32_bf16<<<2304, 256, 0, stream>>>(Wk + wD, wkb, DMODEL * DMODEL);
    cvt_f32_bf16<<<2304, 256, 0, stream>>>(Wv + wD, wvb, DMODEL * DMODEL);
    cvt_f32_bf16<<<2304, 256, 0, stream>>>(Wo + wD, wob, DMODEL * DMODEL);
    cvt_f32_bf16<<<4096, 256, 0, stream>>>(W1 + wF, w1b, DFFN * DMODEL);
    cvt_f32_bf16<<<4096, 256, 0, stream>>>(W2 + wF, w2b, DFFN * DMODEL);

    // Q, K : [B,H,S,dh] ; V stored pre-transposed [B,H,dh,S]
    gemm_nt_bf16<<<gProj, TB, 0, stream>>>(hbf, 0, DMODEL, wqb, 0, DMODEL,
        bq + l * DMODEL, 1.0f, 0, DMODEL, OUT_QK, qb, 0, 0);
    gemm_nt_bf16<<<gProj, TB, 0, stream>>>(hbf, 0, DMODEL, wkb, 0, DMODEL,
        bk + l * DMODEL, 1.0f, 0, DMODEL, OUT_QK, kb2, 0, 0);
    gemm_nt_bf16<<<gProj, TB, 0, stream>>>(hbf, 0, DMODEL, wvb, 0, DMODEL,
        bv + l * DMODEL, 1.0f, 0, DMODEL, OUT_VT, vtb, 0, 0);

    // scores = Q @ K^T * 1/sqrt(dh)   (batched over B*H)
    gemm_nt_bf16<<<gScr, TB, 0, stream>>>(qb, (long long)SEQ * DHEAD, DHEAD,
        kb2, (long long)SEQ * DHEAD, DHEAD,
        nullptr, 0.125f, 0, DHEAD, OUT_BF16, sc, (long long)SEQ * SEQ, SEQ);

    softmax_rows<<<(BATCHN * NHEAD * SEQ) / 4, 128, 0, stream>>>(sc, mask);

    // ctx = attn @ V  (B = V^T, NT form); regathered to [B,S,D]
    gemm_nt_bf16<<<gCtx, TB, 0, stream>>>(sc, (long long)SEQ * SEQ, SEQ,
        vtb, (long long)DHEAD * SEQ, SEQ,
        nullptr, 1.0f, 0, SEQ, OUT_CTX, ctxb, 0, 0);

    // out proj -> fp32, then residual + LN1
    gemm_nt_bf16<<<gProj, TB, 0, stream>>>(ctxb, 0, DMODEL, wob, 0, DMODEL,
        bo + l * DMODEL, 1.0f, 0, DMODEL, OUT_F32, t32, 0, DMODEL);
    add_layernorm<<<NTOK / 4, 128, 0, stream>>>(h32, t32,
        g1 + l * DMODEL, b1 + l * DMODEL, a32, abf);

    // FFN: relu(a @ W1^T + b1) @ W2^T + b2, residual + LN2 -> next h
    gemm_nt_bf16<<<gFfn1, TB, 0, stream>>>(abf, 0, DMODEL, w1b, 0, DMODEL,
        bf1 + l * DFFN, 1.0f, 1, DMODEL, OUT_BF16, f1b, 0, DFFN);
    gemm_nt_bf16<<<gProj, TB, 0, stream>>>(f1b, 0, DFFN, w2b, 0, DFFN,
        bf2 + l * DMODEL, 1.0f, 0, DFFN, OUT_F32, t32, 0, DMODEL);
    add_layernorm<<<NTOK / 4, 128, 0, stream>>>(a32, t32,
        g2 + l * DMODEL, b2 + l * DMODEL, h32, hbf);
  }

  (void)hipMemcpyAsync(d_out, h32, (size_t)NTOK * DMODEL * 4,
                       hipMemcpyDeviceToDevice, stream);
}
